// Model_39676907880817
// MI455X (gfx1250) — matrix operand order repaired and checked by reading, not compiled, not run on hardware
//
#include <hip/hip_runtime.h>
#include <hip/hip_bf16.h>

// ---- problem constants (from reference) ----
#define BB 2
#define HH 16
#define SS 2048
#define DD 64

typedef _Float16 v16h __attribute__((ext_vector_type(16)));
typedef float    v8f  __attribute__((ext_vector_type(8)));
typedef _Float16 h2   __attribute__((ext_vector_type(2)));

static constexpr int MTILE  = 128;  // query rows per block
static constexpr int KTILE  = 32;   // keys per iteration (== WMMA K for f16)
static constexpr int NWAVES = 4;    // 128 threads / wave32; each wave owns 32 q-rows
static constexpr float LOG2E = 1.4426950408889634f;

// v_cvt_pk_rtz_f16_f32 wrapper (builtin returns __fp16x2; bit-identical to h2)
__device__ __forceinline__ h2 cvt_pk(float a, float b) {
    return __builtin_bit_cast(h2, __builtin_amdgcn_cvt_pkrtz(a, b));
}

// lane<->lane^16 exchange (DS pipe, no LDS memory): group-of-32 swizzle SWAPX16
#define SWZ_X16 0x401F
__device__ __forceinline__ float swapx16(float x) {
    return __builtin_bit_cast(float,
        __builtin_amdgcn_ds_swizzle(__builtin_bit_cast(int, x), SWZ_X16));
}

__global__ __launch_bounds__(128)
void fa_fwd_kernel(const float* __restrict__ Q,
                   const float* __restrict__ K,
                   const float* __restrict__ V,
                   const float* __restrict__ scale_p,
                   float* __restrict__ O)
{
    // Carved LDS: main loop uses Ks (8448 B) + Vt (8704 B); the epilogue reuses
    // the whole block as per-wave f32 transpose scratch (4 x 4352 B).
    __shared__ __align__(16) char smem[17408];
    auto Ks = reinterpret_cast<_Float16 (*)[KTILE][DD + 2]>(smem);          // [2][32][66]
    auto Vt = reinterpret_cast<_Float16 (*)[DD][KTILE + 2]>(smem + 8448);   // [2][64][34]

    const int tid  = threadIdx.x;
    const int lane = tid & 31;
    const int wv   = tid >> 5;
    const int hi   = lane >> 4;   // lane half (0/1)
    const int ln   = lane & 15;
    const bool hb  = (hi != 0);

    const int bh = blockIdx.x;                      // 0..B*H-1
    const int q0 = blockIdx.y * MTILE + wv * 32;    // 32 q-rows per wave

    // fold (1/scale)*log2(e) into Q at load time -> scores arrive exp2-ready
    const float kscale = (1.0f / scale_p[0]) * LOG2E;

    const float* Qb = Q + (size_t)bh * SS * DD;
    const float* Kb = K + (size_t)bh * SS * DD;
    const float* Vb = V + (size_t)bh * SS * DD;
    float*       Ob = O + (size_t)bh * SS * DD;

    // ---- staging: 128 threads move one 32x64 f32 tile of K and V ----
    // thread -> key pair (2*(tid>>3), +1), d-span (tid&7)*8 .. +7
    auto stage_load = [&](int kbase, float* kr, float* vr) {
        const int k0 = (tid >> 3) * 2;
        const int dp = (tid & 7) * 8;
#pragma unroll
        for (int p = 0; p < 2; ++p) {
            const float4* ks = (const float4*)(Kb + (size_t)(kbase + k0 + p) * DD + dp);
            const float4* vs = (const float4*)(Vb + (size_t)(kbase + k0 + p) * DD + dp);
#pragma unroll
            for (int i = 0; i < 2; ++i) {
                float4 a = ks[i];
                kr[p*8 + 4*i + 0] = a.x; kr[p*8 + 4*i + 1] = a.y;
                kr[p*8 + 4*i + 2] = a.z; kr[p*8 + 4*i + 3] = a.w;
                float4 b = vs[i];
                vr[p*8 + 4*i + 0] = b.x; vr[p*8 + 4*i + 1] = b.y;
                vr[p*8 + 4*i + 2] = b.z; vr[p*8 + 4*i + 3] = b.w;
            }
        }
    };
    auto stage_store = [&](int buf, const float* kr, const float* vr) {
        const int k0 = (tid >> 3) * 2;
        const int dp = (tid & 7) * 8;
#pragma unroll
        for (int p = 0; p < 2; ++p) {
            h2* kdst = (h2*)&Ks[buf][k0 + p][dp];
#pragma unroll
            for (int i = 0; i < 4; ++i)
                kdst[i] = cvt_pk(kr[p*8 + 2*i], kr[p*8 + 2*i + 1]);
        }
#pragma unroll
        for (int i = 0; i < 8; ++i)                 // transpose: Vt[d][key0..1]
            *(h2*)&Vt[buf][dp + i][k0] = cvt_pk(vr[i], vr[8 + i]);
    };

    // ---- Q as loop-invariant B-fragments (pre-scaled, f32->f16) ----
    // B 16-bit 32x16 layout: lane l col N=l&15 (query); j<4: d=8hi+2j, j>=4: d=16+8hi+2(j-4).
    v16h qb[2][2];
#pragma unroll
    for (int qs = 0; qs < 2; ++qs)
#pragma unroll
        for (int c = 0; c < 2; ++c) {
            const float* qrow = Qb + (size_t)(q0 + qs * 16 + ln) * DD + c * 32;
#pragma unroll
            for (int j = 0; j < 8; ++j) {
                const int d = ((j < 4) ? (2 * j) : (16 + 2 * (j - 4))) + 8 * hi;
                const float2 f = *(const float2*)(qrow + d);
                qb[qs][c][2 * j]     = (_Float16)(f.x * kscale);
                qb[qs][c][2 * j + 1] = (_Float16)(f.y * kscale);
            }
        }

    // ---- state: O^T accumulators; per-lane scalar m/l (lane owns one query) ----
    v8f o[2][4];
#pragma unroll
    for (int q = 0; q < 2; ++q)
#pragma unroll
        for (int t = 0; t < 4; ++t)
#pragma unroll
            for (int i = 0; i < 8; ++i) o[q][t][i] = 0.0f;

    float m_run[2] = { -1.0e30f, -1.0e30f };
    float l_run[2] = { 0.0f, 0.0f };       // lane-partial (own 16 keys per iter)

    // ---- prologue: stage tile 0 into buffer 0 ----
    float kr[16], vr[16];
    stage_load(0, kr, vr);
    stage_store(0, kr, vr);
    __syncthreads();

    int cur = 0;
    for (int kb = 0; kb < SS; kb += KTILE, cur ^= 1) {
        const bool has_next = (kb + KTILE) < SS;   // uniform branch, EXEC stays full
        if (has_next) stage_load(kb + KTILE, kr, vr);

        // ---- S^T = K * Q^T : two key-row tiles, K A-frags reused by both q ----
        // A 16-bit 16x32 layout: lane l row M=l&15 (key); VGPR j<4: k=8hi+2j,
        // j>=4: k=16+8hi+2(j-4).
        v8f sc[2][2];
#pragma unroll
        for (int t = 0; t < 2; ++t) {
            v16h ka0, ka1;
            const _Float16* krow = &Ks[cur][t * 16 + ln][0];
#pragma unroll
            for (int j = 0; j < 8; ++j) {
                const int d = ((j < 4) ? (2 * j) : (16 + 2 * (j - 4))) + 8 * hi;
                const h2 ha = *(const h2*)(krow + d);
                const h2 hc = *(const h2*)(krow + 32 + d);
                ka0[2 * j] = ha[0]; ka0[2 * j + 1] = ha[1];
                ka1[2 * j] = hc[0]; ka1[2 * j + 1] = hc[1];
            }
#pragma unroll
            for (int q = 0; q < 2; ++q) {
                v8f c;
#pragma unroll
                for (int i = 0; i < 8; ++i) c[i] = 0.0f;
                c = __builtin_amdgcn_wmma_f32_16x16x32_f16(
                        false, ka0, false, qb[q][0], (short)0, c, false, false);
                c = __builtin_amdgcn_wmma_f32_16x16x32_f16(
                        false, ka1, false, qb[q][1], (short)0, c, false, false);
                sc[q][t] = c;   // lane: query=l&15; tile t VGPR r: key=t*16+r+8hi
            }
        }

        // ---- softmax per lane-owned query; P^T B-frags built in-register ----
        v16h pb[2];
#pragma unroll
        for (int q = 0; q < 2; ++q) {
            // in-lane max over 16 values, then one xor-16 exchange
            float mx = fmaxf(sc[q][0][0], sc[q][0][1]);
#pragma unroll
            for (int i = 2; i < 8; ++i) mx = fmaxf(mx, sc[q][0][i]);
#pragma unroll
            for (int i = 0; i < 8; ++i) mx = fmaxf(mx, sc[q][1][i]);
            mx = fmaxf(mx, swapx16(mx));
            const float mnew  = fmaxf(m_run[q], mx);
            const float alpha = __builtin_amdgcn_exp2f(m_run[q] - mnew);
            m_run[q] = mnew;

            h2 pkt0[4], pkt1[4];
            float psum = 0.0f;
#pragma unroll
            for (int r = 0; r < 4; ++r) {
                const float a0 = __builtin_amdgcn_exp2f(sc[q][0][2*r]     - mnew);
                const float a1 = __builtin_amdgcn_exp2f(sc[q][0][2*r + 1] - mnew);
                const float b0 = __builtin_amdgcn_exp2f(sc[q][1][2*r]     - mnew);
                const float b1 = __builtin_amdgcn_exp2f(sc[q][1][2*r + 1] - mnew);
                psum += (a0 + a1) + (b0 + b1);
                pkt0[r] = cvt_pk(a0, a1);  // keys 8hi+2r,+1
                pkt1[r] = cvt_pk(b0, b1);  // keys 16+8hi+2r,+1
            }
            l_run[q] = l_run[q] * alpha + psum;
#pragma unroll
            for (int t = 0; t < 4; ++t) o[q][t] *= alpha;      // alpha is per-lane

            // exchange: give partner what it lacks (hi=0 sends pkt1, hi=1 sends pkt0)
#pragma unroll
            for (int r = 0; r < 4; ++r) {
                const int xi = __builtin_bit_cast(int, hb ? pkt0[r] : pkt1[r]);
                const h2 yr = __builtin_bit_cast(h2, __builtin_amdgcn_ds_swizzle(xi, SWZ_X16));
                // B-frag: VGPR j<4 holds keys 16hi+2j,+1 ; j>=4 holds +8..
                const h2 lo = hb ? yr : pkt0[r];   // keys 16hi + 0..7
                const h2 hi4 = hb ? pkt1[r] : yr;  // keys 16hi + 8..15
                pb[q][2 * r]     = lo[0];  pb[q][2 * r + 1]     = lo[1];
                pb[q][8 + 2 * r] = hi4[0]; pb[q][8 + 2 * r + 1] = hi4[1];
            }
        }

        // ---- O^T += V^T * P^T : V A-frags reused by both q-subtiles ----
#pragma unroll
        for (int tt = 0; tt < 4; ++tt) {
            v16h va;
            const _Float16* vrow = &Vt[cur][tt * 16 + ln][0];
#pragma unroll
            for (int j = 0; j < 8; ++j) {
                const int k = 16 * hi + 2 * j;
                const h2 h = *(const h2*)(vrow + k);
                va[2 * j] = h[0]; va[2 * j + 1] = h[1];
            }
#pragma unroll
            for (int q = 0; q < 2; ++q)
                o[q][tt] = __builtin_amdgcn_wmma_f32_16x16x32_f16(
                               false, va, false, pb[q], (short)0, o[q][tt], false, false);
        }

        if (has_next) stage_store(cur ^ 1, kr, vr);
        __syncthreads();
    }

    // ---- epilogue: finish l, normalize, transpose O^T via per-wave LDS ----
    // (final __syncthreads above: all waves are done with Ks/Vt -> safe to reuse)
    float* ow = reinterpret_cast<float*>(smem) + wv * 1088;   // [16 rows][68 f32]
#pragma unroll
    for (int q = 0; q < 2; ++q) {
        const float ltot = l_run[q] + swapx16(l_run[q]);
        const float inv  = __builtin_amdgcn_rcpf(ltot);
        // O^T C-layout: lane = query col (ln), VGPR r of tile tt = d = tt*16+r+8hi
#pragma unroll
        for (int tt = 0; tt < 4; ++tt)
#pragma unroll
            for (int r = 0; r < 8; ++r)
                ow[ln * 68 + tt * 16 + r + 8 * hi] = o[q][tt][r] * inv;
        // same-wave DS ordering guarantees visibility; read back coalesced
        const int qr = lane >> 1;                 // query row 0..15
        const int dblk = (lane & 1) * 32;         // d 0..31 or 32..63
        const float4* src = (const float4*)(ow + qr * 68 + dblk);
        float4* dst = (float4*)(Ob + (size_t)(q0 + q * 16 + qr) * DD + dblk);
#pragma unroll
        for (int i = 0; i < 8; ++i) dst[i] = src[i];
    }
}

extern "C" void kernel_launch(void* const* d_in, const int* in_sizes, int n_in,
                              void* d_out, int out_size, void* d_ws, size_t ws_size,
                              hipStream_t stream)
{
    (void)in_sizes; (void)n_in; (void)out_size; (void)d_ws; (void)ws_size;
    const float* Q     = (const float*)d_in[0];
    const float* K     = (const float*)d_in[1];
    const float* V     = (const float*)d_in[2];
    const float* scale = (const float*)d_in[3];
    // d_in[4] = dropout_p (0) -> ignored, deterministic reference
    float* out = (float*)d_out;

    dim3 grid(BB * HH, SS / MTILE);   // 32 x 16 = 512 blocks, 4 waves each
    fa_fwd_kernel<<<grid, 128, 0, stream>>>(Q, K, V, scale, out);
}
